// TinyCausalSelfAttention_19988777795637
// MI455X (gfx1250) — matrix operand order repaired and checked by reading, not compiled, not run on hardware
//
#include <hip/hip_runtime.h>
#include <stdint.h>

// Problem constants (from reference)
#define DIM   1024
#define NH    16
#define HD    64
#define SEQ   4096
#define BSZ   2
#define MROWS (BSZ*SEQ)                        // 8192
#define PART_STRIDE ((size_t)BSZ*NH*SEQ*HD)    // 8,388,608 elems per q/k part
#define VTD   80                               // HD + 16 (ones/zeros rows)
#define VT_ELEMS ((size_t)BSZ*NH*VTD*SEQ)      // 10,485,760 elems

typedef __attribute__((ext_vector_type(16))) __bf16 v16bf;
typedef __attribute__((ext_vector_type(8)))  float  v8f;

union F8   { v8f v; float f[8]; };
union Frag { v16bf v; uint4 q[2]; };

__device__ __forceinline__ unsigned short f2bf(float x) {
  union { float f; uint32_t u; } c; c.f = x;
  uint32_t r = c.u + 0x7FFFu + ((c.u >> 16) & 1u);
  return (unsigned short)(r >> 16);
}

__device__ __forceinline__ v8f zero8() {
  v8f z = {0.f,0.f,0.f,0.f,0.f,0.f,0.f,0.f};
  return z;
}

// Load a 16-halfword fragment from two 16-byte chunks.
__device__ __forceinline__ v16bf loadFrag(const unsigned short* p0,
                                          const unsigned short* p1) {
  Frag fr;
  fr.q[0] = *(const uint4*)p0;
  fr.q[1] = *(const uint4*)p1;
  return fr.v;
}

// 16-lane-group max reduction via DPP16 butterflies (no LDS traffic):
// quad_perm[1,0,3,2]=xor1, quad_perm[2,3,0,1]=xor2, row_half_mirror=xor7,
// row_mirror=xor15.
template<int CTRL>
__device__ __forceinline__ float dppMaxStep(float x) {
  int xi = __builtin_bit_cast(int, x);
  int yi = __builtin_amdgcn_update_dpp(xi, xi, CTRL, 0xf, 0xf, true);
  return fmaxf(x, __builtin_bit_cast(float, yi));
}
__device__ __forceinline__ float max16(float x) {
  x = dppMaxStep<0x0B1>(x);
  x = dppMaxStep<0x04E>(x);
  x = dppMaxStep<0x141>(x);
  x = dppMaxStep<0x140>(x);
  return x;
}

// ---------------------------------------------------------------------------
// fp32 -> bf16 conversion (4 elems/thread)
// ---------------------------------------------------------------------------
__global__ __launch_bounds__(256) void cvt_f32_to_bf16(
    const float* __restrict__ src, unsigned short* __restrict__ dst, int n4)
{
  int i = blockIdx.x * 256 + threadIdx.x;
  if (i >= n4) return;
  float4 v = ((const float4*)src)[i];
  uint32_t lo = (uint32_t)f2bf(v.x) | ((uint32_t)f2bf(v.y) << 16);
  uint32_t hw = (uint32_t)f2bf(v.z) | ((uint32_t)f2bf(v.w) << 16);
  ((uint2*)dst)[i] = make_uint2(lo, hw);
}

// ---------------------------------------------------------------------------
// Fill V^T rows 64..79 per head: row 64 = ones, rows 65..79 = zeros.
// ---------------------------------------------------------------------------
__global__ __launch_bounds__(256) void init_vt_tail(unsigned short* __restrict__ vt)
{
  int i = blockIdx.x * 256 + threadIdx.x;     // over 32 heads * 16 rows * 4096
  int head = i >> 16;                          // 16*4096 per head
  int rem  = i & 65535;
  int row  = rem >> 12;                        // 0..15
  int s    = rem & 4095;
  vt[((size_t)head * VTD + HD + row) * SEQ + s] = (row == 0) ? 0x3F80 : 0;
}

// ---------------------------------------------------------------------------
// GEMM: C[M,N] = A[M,K] @ W[N,K]^T   (A,W bf16 row-major; K multiple of 32)
// Block: 256 threads (8 waves), tile 128(M) x 128(N); wave tile 32x64.
// Double-buffered LDS with register prefetch; one barrier per K-step.
// EPI==0: scatter bf16 into head-major q (pre-scaled by 1/sqrt(hd)), k, and
//         TRANSPOSED v ([B,H,VTD,S], rows 0..63).   (N = 3*DIM)
// EPI==1: out fp32 = resid + C                      (N = DIM)
// ---------------------------------------------------------------------------
template<int EPI>
__global__ __launch_bounds__(256) void gemm_bf16_wmma(
    const unsigned short* __restrict__ A,
    const unsigned short* __restrict__ W,
    unsigned short* __restrict__ outBf,
    float* __restrict__ outF,
    const float* __restrict__ resid,
    int K)
{
  __shared__ unsigned short sA[2][128 * 40];   // padded stride 40 halfwords
  __shared__ unsigned short sB[2][128 * 40];

  const int tid   = threadIdx.x;
  const int lane  = tid & 31;
  const int wave  = tid >> 5;
  const int waveM = wave >> 1;          // 0..3 -> M offset *32
  const int waveN = wave & 1;           // 0..1 -> N offset *64
  const int mBase = blockIdx.y * 128;
  const int nBase = blockIdx.x * 128;
  const int lm    = lane & 15;
  const int hi    = lane >> 4;

  const int rowT = tid >> 1;            // staging: row 0..127
  const int kcT  = (tid & 1) * 16;      // 0 or 16 halfwords (32B segment)
  const unsigned short* Aptr = A + (size_t)(mBase + rowT) * K + kcT;
  const unsigned short* Wptr = W + (size_t)(nBase + rowT) * K + kcT;

  F8 acc[2][4];
#pragma unroll
  for (int i = 0; i < 2; ++i)
#pragma unroll
    for (int j = 0; j < 4; ++j) acc[i][j].v = zero8();

  // Prologue: stage K-slice 0 into buffer 0
  uint4 pa0 = ((const uint4*)Aptr)[0], pa1 = ((const uint4*)Aptr)[1];
  uint4 pb0 = ((const uint4*)Wptr)[0], pb1 = ((const uint4*)Wptr)[1];
  *(uint4*)(sA[0] + rowT * 40 + kcT)     = pa0;
  *(uint4*)(sA[0] + rowT * 40 + kcT + 8) = pa1;
  *(uint4*)(sB[0] + rowT * 40 + kcT)     = pb0;
  *(uint4*)(sB[0] + rowT * 40 + kcT + 8) = pb1;
  __syncthreads();

  const int nSteps = K >> 5;
  for (int kt = 0; kt < nSteps; ++kt) {
    const int cur  = kt & 1;
    const bool more = (kt + 1) < nSteps;
    if (more) {                         // register prefetch of next K-slice
      const uint4* ap = (const uint4*)(Aptr + (kt + 1) * 32);
      pa0 = ap[0]; pa1 = ap[1];
      const uint4* bp = (const uint4*)(Wptr + (kt + 1) * 32);
      pb0 = bp[0]; pb1 = bp[1];
    }

    v16bf aF[2], bF[4];
#pragma unroll
    for (int mi = 0; mi < 2; ++mi) {
      int row = waveM * 32 + mi * 16 + lm;
      const unsigned short* p = sA[cur] + row * 40 + hi * 8;   // A layout
      aF[mi] = loadFrag(p, p + 16);
    }
#pragma unroll
    for (int ni = 0; ni < 4; ++ni) {
      int row = waveN * 64 + ni * 16 + lm;
      const unsigned short* p = sB[cur] + row * 40 + hi * 8;   // B layout
      bF[ni] = loadFrag(p, p + 16);
    }
#pragma unroll
    for (int mi = 0; mi < 2; ++mi)
#pragma unroll
      for (int ni = 0; ni < 4; ++ni)
        acc[mi][ni].v = __builtin_amdgcn_wmma_f32_16x16x32_bf16(
            false, aF[mi], false, bF[ni], (short)0, acc[mi][ni].v, false, false);

    if (more) {                         // stage prefetched slice into other buf
      *(uint4*)(sA[1 - cur] + rowT * 40 + kcT)     = pa0;
      *(uint4*)(sA[1 - cur] + rowT * 40 + kcT + 8) = pa1;
      *(uint4*)(sB[1 - cur] + rowT * 40 + kcT)     = pb0;
      *(uint4*)(sB[1 - cur] + rowT * 40 + kcT + 8) = pb1;
    }
    __syncthreads();
  }

  // Epilogue (C layout: vgpr r holds M = r + 8*hi, N = lm within 16x16 frag)
#pragma unroll
  for (int mi = 0; mi < 2; ++mi)
#pragma unroll
    for (int ni = 0; ni < 4; ++ni)
#pragma unroll
      for (int r = 0; r < 8; ++r) {
        int m = mBase + waveM * 32 + mi * 16 + hi * 8 + r;
        int n = nBase + waveN * 64 + ni * 16 + lm;
        float val = acc[mi][ni].f[r];
        if (EPI == 0) {
          int part   = n >> 10;          // 0=q 1=k 2=v
          int within = n & 1023;
          int h = within >> 6;
          int c = within & 63;
          int b = m >> 12;
          int s = m & 4095;
          if (part == 0) val *= 0.125f;  // fold 1/sqrt(hd) into Q
          size_t dst;
          if (part < 2)                  // q,k: head-major [B,H,S,hd]
            dst = (size_t)part * PART_STRIDE +
                  ((size_t)(b * NH + h) * SEQ + s) * HD + c;
          else                           // v: transposed [B,H,VTD,S]
            dst = 2 * PART_STRIDE +
                  ((size_t)(b * NH + h) * VTD + c) * SEQ + s;
          outBf[dst] = f2bf(val);
        } else {
          size_t idx = (size_t)m * DIM + n;
          outF[idx] = resid[idx] + val;
        }
      }
}

// ---------------------------------------------------------------------------
// Flash attention (causal). Block = 128 threads (4 waves); each wave owns 32
// q rows of one (b,h), iterates only to its own causal bound, no block
// barriers. Row-sum (flash "l") is computed by the MATRIX pipe via the
// ones-column appended to V^T; row-max uses DPP16 butterflies.
// ---------------------------------------------------------------------------
__global__ __launch_bounds__(128) void flash_attn_wmma(
    const unsigned short* __restrict__ Q,    // [B,H,S,hd], pre-scaled
    const unsigned short* __restrict__ Km,   // [B,H,S,hd]
    const unsigned short* __restrict__ VT,   // [B,H,VTD,S]
    unsigned short* __restrict__ Y)          // [B,S,D]
{
  __shared__ unsigned short sP[4][32 * 40];  // per-wave 32x32 P tile

  const int tid  = threadIdx.x;
  const int lane = tid & 31;
  const int wave = tid >> 5;
  const int lm   = lane & 15;
  const int hi   = lane >> 4;
  const int qblk = blockIdx.x;
  const int bh   = blockIdx.y;
  const int b    = bh >> 4;
  const int h    = bh & 15;

  const unsigned short* Qh  = Q  + (size_t)bh * SEQ * HD;
  const unsigned short* Kh  = Km + (size_t)bh * SEQ * HD;
  const unsigned short* VTh = VT + (size_t)bh * VTD * SEQ;

  const int qBase = qblk * 128 + wave * 32;

  // Q fragments (A layout): 2 m-frags x 2 k-frags, direct from global
  v16bf qF[2][2];
#pragma unroll
  for (int mi = 0; mi < 2; ++mi)
#pragma unroll
    for (int kf = 0; kf < 2; ++kf) {
      const unsigned short* p =
          Qh + (size_t)(qBase + mi * 16 + lm) * HD + kf * 32 + hi * 8;
      qF[mi][kf] = loadFrag(p, p + 16);
    }

  // o[mi][0..3] = output cols; o[mi][4] = flash "l" (ones-column of V^T)
  F8 o[2][5];
#pragma unroll
  for (int mi = 0; mi < 2; ++mi)
#pragma unroll
    for (int of = 0; of < 5; ++of) o[mi][of].v = zero8();
  float mrow[2][8];
#pragma unroll
  for (int mi = 0; mi < 2; ++mi)
#pragma unroll
    for (int r = 0; r < 8; ++r) mrow[mi][r] = -1e30f;

  const int ktCount = qBase / 32 + 1;        // exact causal bound per wave
  unsigned short* pB = sP[wave];

  for (int kt = 0; kt < ktCount; ++kt) {
    const int kBase = kt * 32;

    if (kt + 1 < ktCount) {                  // prefetch next K / V^T tiles
      __builtin_prefetch(Kh + (size_t)(kBase + 32 + lane) * HD, 0, 3);
      __builtin_prefetch(VTh + (size_t)(lane << 1) * SEQ + kBase + 32, 0, 3);
    }

    // S = Q @ K^T for 32 keys: K frags shared by both m-frags
    v16bf kF[2][2];
#pragma unroll
    for (int nf = 0; nf < 2; ++nf)
#pragma unroll
      for (int kf = 0; kf < 2; ++kf) {
        const unsigned short* p =
            Kh + (size_t)(kBase + nf * 16 + lm) * HD + kf * 32 + hi * 8;
        kF[nf][kf] = loadFrag(p, p + 16);    // B layout, table order per lane
      }
    F8 sAcc[2][2];
#pragma unroll
    for (int mi = 0; mi < 2; ++mi)
#pragma unroll
      for (int nf = 0; nf < 2; ++nf) {
        sAcc[mi][nf].v = zero8();
#pragma unroll
        for (int kf = 0; kf < 2; ++kf)
          sAcc[mi][nf].v = __builtin_amdgcn_wmma_f32_16x16x32_bf16(
              false, qF[mi][kf], false, kF[nf][kf], (short)0,
              sAcc[mi][nf].v, false, false);
      }

    // Causal mask: only the diagonal tile needs it (wave-uniform branch)
    if (kBase + 32 > qBase) {
#pragma unroll
      for (int mi = 0; mi < 2; ++mi)
#pragma unroll
        for (int nf = 0; nf < 2; ++nf)
#pragma unroll
          for (int r = 0; r < 8; ++r) {
            int qRow = qBase + mi * 16 + hi * 8 + r;
            int kIdx = kBase + nf * 16 + lm;
            if (kIdx > qRow) sAcc[mi][nf].f[r] = -1e30f;
          }
    }

    // Online softmax: DPP16 max butterfly; exp fused straight into the
    // per-wave LDS P tile (C layout -> A layout shuffle).
#pragma unroll
    for (int mi = 0; mi < 2; ++mi)
#pragma unroll
      for (int r = 0; r < 8; ++r) {
        float tmax = max16(fmaxf(sAcc[mi][0].f[r], sAcc[mi][1].f[r]));
        float mnew  = fmaxf(mrow[mi][r], tmax);
        float alpha = __expf(mrow[mi][r] - mnew);
        mrow[mi][r] = mnew;
        int rowOff = (mi * 16 + hi * 8 + r) * 40;
        pB[rowOff + lm]      = f2bf(__expf(sAcc[mi][0].f[r] - mnew));
        pB[rowOff + 16 + lm] = f2bf(__expf(sAcc[mi][1].f[r] - mnew));
#pragma unroll
        for (int of = 0; of < 5; ++of) o[mi][of].f[r] *= alpha;
      }

    asm volatile("s_wait_dscnt 0" ::: "memory");

    v16bf pF[2];
#pragma unroll
    for (int mi = 0; mi < 2; ++mi) {
      const unsigned short* pp = pB + (mi * 16 + lm) * 40 + hi * 8;
      pF[mi] = loadFrag(pp, pp + 16);
    }

    // O += P @ [V | 1 | 0..] : V^T fragments direct from global; the 5th
    // fragment (rows 64..79 of V^T) accumulates the row-sum "l".
#pragma unroll
    for (int of = 0; of < 5; ++of) {
      const unsigned short* vp =
          VTh + (size_t)(of * 16 + lm) * SEQ + kBase + hi * 8;
      v16bf vF = loadFrag(vp, vp + 16);
#pragma unroll
      for (int mi = 0; mi < 2; ++mi)
        o[mi][of].v = __builtin_amdgcn_wmma_f32_16x16x32_bf16(
            false, pF[mi], false, vF, (short)0, o[mi][of].v, false, false);
    }
  }

  // Normalize (l lives in o[mi][4] column 0 of each 16-lane group) and write
  // y (merged-head [B,S,D] bf16)
#pragma unroll
  for (int mi = 0; mi < 2; ++mi) {
    float linv[8];
#pragma unroll
    for (int r = 0; r < 8; ++r)
      linv[r] = 1.0f / __shfl(o[mi][4].f[r], lane & 16, 32);
#pragma unroll
    for (int of = 0; of < 4; ++of)
#pragma unroll
      for (int r = 0; r < 8; ++r) {
        int qRow  = qBase + mi * 16 + hi * 8 + r;
        float val = o[mi][of].f[r] * linv[r];
        size_t idx = ((size_t)b * SEQ + qRow) * DIM + h * HD + of * 16 + lm;
        Y[idx] = f2bf(val);
      }
  }
}

// ---------------------------------------------------------------------------
// Launch. Workspace layout (bf16 elems):
//   xbf 8.4M | wqkv 3.1M | wproj 1.0M | q 8.4M | k 8.4M | vT 10.5M | y 8.4M
//   total ~= 96.6 MB
// ---------------------------------------------------------------------------
extern "C" void kernel_launch(void* const* d_in, const int* in_sizes, int n_in,
                              void* d_out, int out_size, void* d_ws, size_t ws_size,
                              hipStream_t stream) {
  const float* x     = (const float*)d_in[0];
  const float* Wqkv  = (const float*)d_in[1];
  const float* Wproj = (const float*)d_in[2];
  float* out = (float*)d_out;

  unsigned short* xbf     = (unsigned short*)d_ws;
  unsigned short* wqkvbf  = xbf     + (size_t)MROWS * DIM;
  unsigned short* wprojbf = wqkvbf  + (size_t)3 * DIM * DIM;
  unsigned short* qkv     = wprojbf + (size_t)DIM * DIM;   // q | k | vT
  unsigned short* ybf     = qkv + 2 * PART_STRIDE + VT_ELEMS;

  cvt_f32_to_bf16<<<(MROWS * DIM / 4) / 256, 256, 0, stream>>>(x, xbf, MROWS * DIM / 4);
  cvt_f32_to_bf16<<<(3 * DIM * DIM / 4) / 256, 256, 0, stream>>>(Wqkv, wqkvbf, 3 * DIM * DIM / 4);
  cvt_f32_to_bf16<<<(DIM * DIM / 4) / 256, 256, 0, stream>>>(Wproj, wprojbf, DIM * DIM / 4);
  init_vt_tail<<<(BSZ * NH * 16 * SEQ) / 256, 256, 0, stream>>>(qkv + 2 * PART_STRIDE);

  // QKV projection -> head-major q (pre-scaled), k, transposed v
  gemm_bf16_wmma<0><<<dim3(3 * DIM / 128, MROWS / 128), 256, 0, stream>>>(
      xbf, wqkvbf, qkv, nullptr, nullptr, DIM);

  // Causal flash attention (barrier-free inner loop)
  flash_attn_wmma<<<dim3(SEQ / 128, BSZ * NH), 128, 0, stream>>>(
      qkv, qkv + PART_STRIDE, qkv + 2 * PART_STRIDE, ybf);

  // Output projection + residual
  gemm_bf16_wmma<1><<<dim3(DIM / 128, MROWS / 128), 256, 0, stream>>>(
      ybf, wprojbf, nullptr, out, x, DIM);
}
